// TDTFLayer_23141283791225
// MI455X (gfx1250) — matrix operand order repaired and checked by reading, not compiled, not run on hardware
//
#include <hip/hip_runtime.h>
#include <cstdint>
#include <cstddef>

// Problem constants (fixed by the reference)
#define D_DIM 2048
#define I_DIM 5504
#define B_TOK 256
#define KSEL  128
#define EPSV  1e-6f

typedef __attribute__((ext_vector_type(16))) __bf16 v16bf;
typedef __attribute__((ext_vector_type(8)))  float  v8f;
typedef __attribute__((ext_vector_type(2)))  float  f32x2;
typedef int vi4 __attribute__((__vector_size__(16)));

#if defined(__has_builtin)
#  if __has_builtin(__builtin_amdgcn_global_load_async_to_lds_b128)
#    define ASYNC_BUILTIN 1
#  else
#    define ASYNC_BUILTIN 0
#  endif
#  if __has_builtin(__builtin_amdgcn_s_wait_asynccnt)
#    define WAITASYNC_BUILTIN 1
#  else
#    define WAITASYNC_BUILTIN 0
#  endif
#else
#  define ASYNC_BUILTIN 0
#  define WAITASYNC_BUILTIN 0
#endif

// HBM -> LDS async copy (ASYNCcnt; decoupled from LOADcnt)
__device__ __forceinline__ void async_to_lds_b128(const unsigned short* g,
                                                  unsigned short* l) {
#if ASYNC_BUILTIN
    __builtin_amdgcn_global_load_async_to_lds_b128(
        (__attribute__((address_space(1))) vi4*)(unsigned short*)g,
        (__attribute__((address_space(3))) vi4*)l, 0, 0);
#else
    unsigned lds = (unsigned)(unsigned long long)
                   (__attribute__((address_space(3))) void*)l;
    asm volatile("global_load_async_to_lds_b128 %0, %1, off"
                 :: "v"(lds), "v"((unsigned long long)g)
                 : "memory");
#endif
}

__device__ __forceinline__ void wait_async0() {
#if WAITASYNC_BUILTIN
    __builtin_amdgcn_s_wait_asynccnt(0);
#else
    asm volatile("s_wait_asynccnt 0x0" ::: "memory");
#endif
}

// fp32 -> bf16, round-to-nearest-even
__device__ __forceinline__ unsigned short f2bf(float x) {
    union { float f; unsigned u; } c; c.f = x;
    unsigned r = c.u + 0x7FFFu + ((c.u >> 16) & 1u);
    return (unsigned short)(r >> 16);
}

// ---------------------------------------------------------------- copy out
__global__ void copy_f32_kernel(const float4* __restrict__ src,
                                float4* __restrict__ dst, int n4) {
    int i = blockIdx.x * blockDim.x + threadIdx.x;
    if (i < n4) dst[i] = src[i];
}

// ---------------------------------------------------------------- router scores
__global__ void router_kernel(const float* __restrict__ hs,
                              const float* __restrict__ rw,
                              const float* __restrict__ rb,
                              float* __restrict__ scores) {
    __shared__ float red[256];
    const int b = blockIdx.x;
    const float* row = hs + (size_t)b * D_DIM;
    float s = 0.f;
    for (int i = threadIdx.x; i < D_DIM; i += 256) s += row[i] * rw[i];
    red[threadIdx.x] = s;
    __syncthreads();
    for (int off = 128; off > 0; off >>= 1) {
        if (threadIdx.x < off) red[threadIdx.x] += red[threadIdx.x + off];
        __syncthreads();
    }
    if (threadIdx.x == 0) scores[b] = red[0] + rb[0];
}

// ---------------------------------------------------------------- exact top-k by rank
// sigmoid is monotone -> rank raw scores. Tie-break: smaller index wins
// (matches jax.lax.top_k stability). Ranks form a total order -> unique slots.
__global__ void select_kernel(const float* __restrict__ scores,
                              int* __restrict__ sel) {
    __shared__ float s[B_TOK];
    const int t = threadIdx.x;
    s[t] = scores[t];
    __syncthreads();
    const float mine = s[t];
    int rank = 0;
    for (int j = 0; j < B_TOK; ++j) {
        float v = s[j];
        rank += (v > mine) || (v == mine && j < t);
    }
    if (rank < KSEL) sel[rank] = t;
}

// ---------------------------------------------------------------- RMSNorm -> bf16
template<bool GATHER>
__global__ void rms_kernel(const float* __restrict__ x,
                           const int* __restrict__ sel,
                           const float* __restrict__ w,
                           unsigned short* __restrict__ h) {
    __shared__ float red[256];
    __shared__ float inv;
    const int row = blockIdx.x;
    const int tid = threadIdx.x;
    const float* xr = x + (size_t)(GATHER ? sel[row] : row) * D_DIM;
    float s = 0.f;
    for (int i = tid; i < D_DIM; i += 256) { float v = xr[i]; s += v * v; }
    red[tid] = s;
    __syncthreads();
    for (int off = 128; off > 0; off >>= 1) {
        if (tid < off) red[tid] += red[tid + off];
        __syncthreads();
    }
    if (tid == 0) inv = rsqrtf(red[0] * (1.f / D_DIM) + EPSV);
    __syncthreads();
    const float k = inv;
    for (int i = tid; i < D_DIM; i += 256)
        h[(size_t)row * D_DIM + i] = f2bf(xr[i] * k * w[i]);
}

// ---------------------------------------------------------------- WMMA GEMM, M=128
// C[128,N] = A_bf16[128,K] @ B_f32[K,N]  (+ fused epilogue per MODE)
// 256 threads = 8 waves. Wave w -> M rows [16w,16w+16). Block -> 16*NSUB N cols.
//
// Counter-decoupled pipeline per K-step (one barrier):
//  - A tile (bf16, 8KB) : HBM -> LDS via ASYNC loads (ASYNCcnt), waited with
//    s_wait_asynccnt right before the barrier -> a full K-step of slack.
//  - B tile (fp32)      : NT global -> regs (LOADcnt), converted to bf16 and
//    ds_store'd at the bottom of the iteration, latency covered by the WMMAs.
//  - WMMA reads both fragments from LDS, so its wait is DScnt only; it can
//    never be stalled on the streaming weight loads.
enum { MODE_V = 0, MODE_ATTN = 1, MODE_GATEUP = 2, MODE_DOWN = 3 };

template<int MODE, int KDIM, int NDIM, int NSUB>
__global__ __launch_bounds__(256)
void gemm_kernel(const unsigned short* __restrict__ A,   // [128][KDIM] bf16 bits
                 const float* __restrict__ B0,           // [KDIM][NDIM]
                 const float* __restrict__ B1,           // gate/up second weight
                 const float* __restrict__ bias,         // MODE_V
                 const float* __restrict__ x1,           // MODE_DOWN residual [128][2048]
                 const float* __restrict__ hidden,       // MODE_ATTN residual source
                 const int*   __restrict__ sel,          // MODE_ATTN / MODE_DOWN
                 unsigned short* __restrict__ outbf,     // bf16 output
                 float* __restrict__ outf)               // f32 output
{
    constexpr int NW = (MODE == MODE_GATEUP) ? 2 : 1;
    constexpr int NT = 16 * NSUB;
    // A tile: row-major [row][k] so a lane's A-frag is 2 contiguous 16B reads.
    __shared__ __align__(16) unsigned short As[2][128][32];
    // B tile: K-major per column so a lane's B-frag is one contiguous 32B read.
    __shared__ __align__(32) unsigned short Bs[NW][2][NT][32];

    const int tid   = threadIdx.x;
    const int lane  = tid & 31;
    const int wave  = tid >> 5;
    const int hf    = lane >> 4;     // lane half selects K-subset (A) / K-block (B)
    const int l16   = lane & 15;
    const int nbase = blockIdx.x * NT;
    const int arow  = wave * 16 + l16;

    v8f zero = {0.f,0.f,0.f,0.f,0.f,0.f,0.f,0.f};
    v8f acc[NW][NSUB];
#pragma unroll
    for (int w = 0; w < NW; ++w)
#pragma unroll
        for (int s = 0; s < NSUB; ++s) acc[w][s] = zero;

    // Cooperative B-tile mapping: thread owns a 2(K) x NSUB(col) patch.
    const int p     = tid >> 4;          // 0..15 -> K rows 2p, 2p+1
    const int cbase = (tid & 15) * NSUB; // starting column
    // Cooperative A-tile mapping: thread owns row ar, 16-element K segment.
    const int ar    = tid >> 1;          // 0..127
    const int aseg  = (tid & 1) * 16;    // 0 or 16

    const float* Bp[2] = { B0, B1 };
    float r0[NW][NSUB], r1[NW][NSUB];

    auto issue_asyncA = [&](int k0, int bsel) {   // HBM -> LDS, ASYNCcnt
        const unsigned short* g = A + (size_t)ar * KDIM + k0 + aseg;
        unsigned short* l = &As[bsel][ar][aseg];
        async_to_lds_b128(g, l);
        async_to_lds_b128(g + 8, l + 8);
    };
    auto load_tile = [&](int k0) {       // global -> regs, non-temporal
#pragma unroll
        for (int w = 0; w < NW; ++w) {
            const float* base = Bp[w] + (size_t)(k0 + 2 * p) * NDIM + nbase + cbase;
            if constexpr (NSUB == 2) {
                f32x2 a = __builtin_nontemporal_load((const f32x2*)base);
                f32x2 b = __builtin_nontemporal_load((const f32x2*)(base + NDIM));
                r0[w][0] = a.x; r0[w][1] = a.y;
                r1[w][0] = b.x; r1[w][1] = b.y;
            } else {
                r0[w][0] = __builtin_nontemporal_load(base);
                r1[w][0] = __builtin_nontemporal_load(base + NDIM);
            }
        }
    };
    auto store_tile = [&](int bsel) {    // regs -> bf16 -> LDS (b32 stores)
#pragma unroll
        for (int w = 0; w < NW; ++w)
#pragma unroll
            for (int c = 0; c < NSUB; ++c) {
                unsigned u = (unsigned)f2bf(r0[w][c]) |
                             ((unsigned)f2bf(r1[w][c]) << 16);
                *(unsigned*)&Bs[w][bsel][cbase + c][2 * p] = u;
            }
    };
    auto compute = [&](int bsel) {       // frags from LDS only -> DScnt wait
        // A frag in contiguous K order (matches B): lanes 0-15 hold K {0..15},
        // lanes 16-31 hold K {16..31}
        v16bf afrag;
        ((uint4*)&afrag)[0] = *(const uint4*)&As[bsel][arow][hf * 16];
        ((uint4*)&afrag)[1] = *(const uint4*)&As[bsel][arow][hf * 16 + 8];
#pragma unroll
        for (int s = 0; s < NSUB; ++s)
#pragma unroll
            for (int w = 0; w < NW; ++w) {
                v16bf bfrag = *(const v16bf*)&Bs[w][bsel][s * 16 + l16][hf * 16];
                acc[w][s] = __builtin_amdgcn_wmma_f32_16x16x32_bf16(
                                false, afrag, false, bfrag, (short)0,
                                acc[w][s], false, false);
            }
    };

    issue_asyncA(0, 0);
    load_tile(0);
    store_tile(0);
    int buf = 0;

    for (int k0 = 0; k0 < KDIM - 32; k0 += 32) {
        wait_async0();                         // A tile for `buf` has landed
        __syncthreads();                       // publish A+B tiles for `buf`
        issue_asyncA(k0 + 32, buf ^ 1);        // next A tile (ASYNCcnt)
        load_tile(k0 + 32);                    // next B tile -> regs (LOADcnt)
        compute(buf);                          // WMMAs wait on DScnt only
        store_tile(buf ^ 1);                   // convert+fill other buffer
        buf ^= 1;
    }
    wait_async0();                             // peeled last K-step
    __syncthreads();
    compute(buf);

    // Epilogue. C layout: VGPR r, lane -> M = wave*16 + hf*8 + r, N = ncol.
#pragma unroll
    for (int s = 0; s < NSUB; ++s) {
        const int n = nbase + s * 16 + l16;
#pragma unroll
        for (int r = 0; r < 8; ++r) {
            const int m = wave * 16 + hf * 8 + r;
            const float v = acc[0][s][r];
            if constexpr (MODE == MODE_V) {
                outbf[(size_t)m * NDIM + n] = f2bf(v + bias[n]);
            } else if constexpr (MODE == MODE_ATTN) {
                const float xo = hidden[(size_t)sel[m] * D_DIM + n];
                outf[(size_t)m * NDIM + n] = xo + v;
            } else if constexpr (MODE == MODE_GATEUP) {
                const float g = v, u = acc[NW - 1][s][r];
                const float sg = g / (1.f + __expf(-g));   // silu(g)
                outbf[(size_t)m * NDIM + n] = f2bf(sg * u);
            } else { // MODE_DOWN: scatter x1 + mlp into the full output
                const float xr = x1[(size_t)m * D_DIM + n];
                outf[(size_t)sel[m] * D_DIM + n] = xr + v;
            }
        }
    }
}

// ---------------------------------------------------------------- launch
extern "C" void kernel_launch(void* const* d_in, const int* in_sizes, int n_in,
                              void* d_out, int out_size, void* d_ws, size_t ws_size,
                              hipStream_t stream) {
    const float* hidden   = (const float*)d_in[0];
    // d_in[1]=cos, d_in[2]=sin : unused (T=1 -> softmax==1 -> o==v exactly)
    const float* router_w = (const float*)d_in[3];
    const float* router_b = (const float*)d_in[4];
    const float* ln1_w    = (const float*)d_in[5];
    const float* ln2_w    = (const float*)d_in[6];
    // d_in[7..10] = wq,bq,wk,bk : unused for the same reason
    const float* wv       = (const float*)d_in[11];
    const float* bv       = (const float*)d_in[12];
    const float* wo       = (const float*)d_in[13];
    const float* w_gate   = (const float*)d_in[14];
    const float* w_up     = (const float*)d_in[15];
    const float* w_down   = (const float*)d_in[16];
    float* out = (float*)d_out;

    char* ws = (char*)d_ws;
    float* scores        = (float*)ws;          ws += 1024;
    int*   sel           = (int*)ws;            ws += 1024;
    unsigned short* h1   = (unsigned short*)ws; ws += (size_t)KSEL * D_DIM * 2;
    unsigned short* vbuf = (unsigned short*)ws; ws += (size_t)KSEL * D_DIM * 2;
    float* x1            = (float*)ws;          ws += (size_t)KSEL * D_DIM * 4;
    unsigned short* h2   = (unsigned short*)ws; ws += (size_t)KSEL * D_DIM * 2;
    unsigned short* mbuf = (unsigned short*)ws; ws += (size_t)KSEL * I_DIM * 2;

    // 0) out = hidden_states (selected rows overwritten by MODE_DOWN scatter)
    {
        const int n4 = B_TOK * D_DIM / 4;
        copy_f32_kernel<<<(n4 + 255) / 256, 256, 0, stream>>>(
            (const float4*)hidden, (float4*)out, n4);
    }
    // 1) router scores, 2) exact top-128 selection
    router_kernel<<<B_TOK, 256, 0, stream>>>(hidden, router_w, router_b, scores);
    select_kernel<<<1, 256, 0, stream>>>(scores, sel);
    // 3) h1 = rms(gather(x)) in bf16
    rms_kernel<true><<<KSEL, 256, 0, stream>>>(hidden, sel, ln1_w, h1);
    // 4) v = h1 @ wv + bv  (bf16)
    gemm_kernel<MODE_V, D_DIM, D_DIM, 1><<<D_DIM / 16, 256, 0, stream>>>(
        h1, wv, nullptr, bv, nullptr, nullptr, nullptr, vbuf, nullptr);
    // 5) x1 = gather(x) + v @ wo  (f32)
    gemm_kernel<MODE_ATTN, D_DIM, D_DIM, 1><<<D_DIM / 16, 256, 0, stream>>>(
        vbuf, wo, nullptr, nullptr, nullptr, hidden, sel, nullptr, x1);
    // 6) h2 = rms(x1) in bf16
    rms_kernel<false><<<KSEL, 256, 0, stream>>>(x1, nullptr, ln2_w, h2);
    // 7) m = silu(h2 @ w_gate) * (h2 @ w_up)  (bf16, 32-wide N tile)
    gemm_kernel<MODE_GATEUP, D_DIM, I_DIM, 2><<<I_DIM / 32, 256, 0, stream>>>(
        h2, w_gate, w_up, nullptr, nullptr, nullptr, nullptr, mbuf, nullptr);
    // 8) out[sel] = x1 + m @ w_down
    gemm_kernel<MODE_DOWN, I_DIM, D_DIM, 1><<<D_DIM / 16, 256, 0, stream>>>(
        mbuf, w_down, nullptr, nullptr, x1, nullptr, sel, nullptr, out);

    (void)in_sizes; (void)n_in; (void)out_size; (void)ws_size;
}
